// Fea2GS_20942260535584
// MI455X (gfx1250) — matrix operand order repaired and checked by reading, not compiled, not run on hardware
//
#include <hip/hip_runtime.h>

// ---------------- problem constants ----------------
#define DIMF 180
#define DP   192          // feature dim padded to multiple of 32
#define NTOK 144          // tokens per window
#define NWIN 576          // B * HCOUNT * WCOUNT
#define MTOT (NWIN * NTOK) // 82944 rows
#define NHEAD 6
#define HD 30
#define GSQ 12
#define SIDE 144          // 12 windows * 12 tokens
#define SHIFT 6
#define KPAD 160          // 144 padded to multiple of 32
#define CH 64             // windows per attention chunk
#define NCHUNK 9
#define MKV 2304          // NWIN * 4 rows (scale embedding)

typedef __attribute__((ext_vector_type(16))) __bf16 bf16x16;
typedef __attribute__((ext_vector_type(8)))  float  f32x8;

__device__ __forceinline__ unsigned short f2bf(float f) {
  unsigned int u = __float_as_uint(f);
  return (unsigned short)((u + 0x7FFFu + ((u >> 16) & 1u)) >> 16);
}

// A fragment (16x32 bf16): lane L -> m = L&15, half = L>>4.
// VGPR v: K = 16*half + 2v.
__device__ __forceinline__ bf16x16 load_a_frag(const unsigned short* p, int half) {
  union { bf16x16 v; unsigned int u[8]; } f;
#pragma unroll
  for (int i = 0; i < 8; ++i) f.u[i] = *(const unsigned int*)(p + 16 * half + 2 * i);
  return f.v;
}
// B fragment (32x16 bf16): lane L -> n = L&15, half = L>>4.
// VGPR v: K = 16*half + 2v. p points at B-row n (consecutive K).
__device__ __forceinline__ bf16x16 load_b_frag(const unsigned short* p, int half) {
  union { bf16x16 v; unsigned int u[8]; } f;
#pragma unroll
  for (int i = 0; i < 8; ++i) f.u[i] = *(const unsigned int*)(p + 16 * half + 2 * i);
  return f.v;
}
__device__ __forceinline__ f32x8 wmma_bf16(bf16x16 a, bf16x16 b, f32x8 c) {
  return __builtin_amdgcn_wmma_f32_16x16x32_bf16(false, a, false, b, (short)0, c, false, false);
}

// ---------------- LayerNorm (wave per row, optional post-LN add) ----------------
__global__ void ln_k(const float* __restrict__ x, const float* __restrict__ w,
                     const float* __restrict__ b, const float* __restrict__ addafter,
                     float* __restrict__ out, int rows) {
  int row = blockIdx.x * (blockDim.x >> 5) + (threadIdx.x >> 5);
  if (row >= rows) return;
  int lane = threadIdx.x & 31;
  const float* xr = x + (size_t)row * DIMF;
  float vals[6];
  float s = 0.f;
#pragma unroll
  for (int i = 0; i < 6; ++i) {
    int c = lane + 32 * i;
    float v = (c < DIMF) ? xr[c] : 0.f;
    vals[i] = v; s += v;
  }
#pragma unroll
  for (int m = 16; m; m >>= 1) s += __shfl_xor(s, m, 32);
  float mu = s * (1.f / DIMF);
  float vs = 0.f;
#pragma unroll
  for (int i = 0; i < 6; ++i) {
    int c = lane + 32 * i;
    float d = (c < DIMF) ? (vals[i] - mu) : 0.f;
    vs += d * d;
  }
#pragma unroll
  for (int m = 16; m; m >>= 1) vs += __shfl_xor(vs, m, 32);
  float rstd = rsqrtf(vs * (1.f / DIMF) + 1e-5f);
  float* orow = out + (size_t)row * DIMF;
#pragma unroll
  for (int i = 0; i < 6; ++i) {
    int c = lane + 32 * i;
    if (c < DIMF) {
      float v = (vals[i] - mu) * rstd * w[c] + b[c];
      if (addafter) v += addafter[(size_t)row * DIMF + c];
      orow[c] = v;
    }
  }
}

// ---------------- conversions ----------------
__global__ void cvtA_k(const float* __restrict__ in, unsigned short* __restrict__ out, int rows) {
  size_t i = (size_t)blockIdx.x * blockDim.x + threadIdx.x;
  size_t total = (size_t)rows * DP;
  if (i >= total) return;
  int c = (int)(i % DP);
  size_t r = i / DP;
  float v = (c < DIMF) ? in[r * DIMF + c] : 0.f;
  out[i] = f2bf(v);
}
__global__ void cvtW_k(const float* __restrict__ W, unsigned short* __restrict__ out) {
  int i = blockIdx.x * blockDim.x + threadIdx.x;
  if (i >= DP * DP) return;
  int n = i / DP, k = i % DP;
  float v = (n < DIMF && k < DIMF) ? W[n * DIMF + k] : 0.f;
  out[i] = f2bf(v);
}

// -------- generic GEMM: C(M,180) = A(M,192)bf16 @ Bw(192,192)bf16^T --------
// Register-blocked 1x4: each wave computes one 16-row tile x four 16-col tiles,
// reusing the A fragment across 4 WMMAs per K step (24 WMMAs per wave).
// Optional f32 output (C, stride 180, with bias/residual/relu) and optional
// bf16 padded output (Cbf, stride 192, pad cols zeroed) for chained GEMMs.
__global__ void gemm_wmma(const unsigned short* __restrict__ A,
                          const unsigned short* __restrict__ Bw,
                          const float* __restrict__ bias,
                          const float* __restrict__ residual,
                          float* __restrict__ C,
                          unsigned short* __restrict__ Cbf,
                          int Mrows, int relu) {
  int lane = threadIdx.x & 31;
  int wtile = blockIdx.x * (blockDim.x >> 5) + (threadIdx.x >> 5);
  int rowTiles = Mrows >> 4;
  if (wtile >= rowTiles * 3) return;      // wave-uniform exit
  int rt = wtile / 3, cg = wtile % 3;     // cg selects 4 col tiles (64 cols)
  int mrow = rt << 4, ncol0 = cg * 64;
  int half = lane >> 4, mn = lane & 15;
  const unsigned short* aRow = A + (size_t)(mrow + mn) * DP;
  const unsigned short* bRow[4];
#pragma unroll
  for (int t = 0; t < 4; ++t) bRow[t] = Bw + (size_t)(ncol0 + t * 16 + mn) * DP;
  f32x8 acc[4] = {};
#pragma unroll
  for (int kk = 0; kk < 6; ++kk) {
    if (kk < 5) __builtin_prefetch(aRow + (kk + 1) * 32, 0, 1);
    bf16x16 a = load_a_frag(aRow + kk * 32, half);
#pragma unroll
    for (int t = 0; t < 4; ++t)
      acc[t] = wmma_bf16(a, load_b_frag(bRow[t] + kk * 32, half), acc[t]);
  }
#pragma unroll
  for (int t = 0; t < 4; ++t) {
    int col = ncol0 + t * 16 + mn;
    float bv = (col < DIMF && bias) ? bias[col] : 0.f;
#pragma unroll
    for (int v = 0; v < 8; ++v) {
      int row = mrow + v + 8 * half;
      float val = acc[t][v] + bv;
      if (col < DIMF) {
        if (residual) val += residual[(size_t)row * DIMF + col];
        if (relu) val = fmaxf(val, 0.f);
        if (C) C[(size_t)row * DIMF + col] = val;
      }
      if (Cbf) Cbf[(size_t)row * DP + col] = f2bf(col < DIMF ? val : 0.f);
    }
  }
}

// ---------------- head-layout staging ----------------
// dst[(wl*6+h)*144 + t][d<32] = src[(b0+wl)*144+t][h*30+d] * scale (pad d>=30 -> 0)
__global__ void cvt_head_k(const float* __restrict__ src, unsigned short* __restrict__ dst,
                           float scale, int nwin, int b0) {
  size_t i = (size_t)blockIdx.x * blockDim.x + threadIdx.x;
  size_t total = (size_t)nwin * NHEAD * NTOK * 32;
  if (i >= total) return;
  int d = (int)(i & 31);
  size_t j = i >> 5;
  int t = (int)(j % NTOK); j /= NTOK;
  int h = (int)(j % NHEAD);
  int wl = (int)(j / NHEAD);
  float v = 0.f;
  if (d < HD) v = src[((size_t)(b0 + wl) * NTOK + t) * DIMF + h * HD + d] * scale;
  dst[i] = f2bf(v);
}
// V transposed: dst[(wl*6+h)*32 + d][t<160] = src[...t][h*30+d]  (pad d>=30 or t>=144 -> 0)
__global__ void cvt_vt_k(const float* __restrict__ src, unsigned short* __restrict__ dst,
                         int nwin, int b0) {
  size_t i = (size_t)blockIdx.x * blockDim.x + threadIdx.x;
  size_t total = (size_t)nwin * NHEAD * 32 * KPAD;
  if (i >= total) return;
  int t = (int)(i % KPAD);
  size_t j = i / KPAD;
  int d = (int)(j % 32); j /= 32;
  int h = (int)(j % NHEAD);
  int wl = (int)(j / NHEAD);
  float v = 0.f;
  if (d < HD && t < NTOK) v = src[((size_t)(b0 + wl) * NTOK + t) * DIMF + h * HD + d];
  dst[i] = f2bf(v);
}
// cross-attn K: dst[(b*6+h)*16 + j][d<32] = kf[(b*4+j)*180 + h*30+d]
__global__ void cvt_ckb_k(const float* __restrict__ kf, unsigned short* __restrict__ dst) {
  size_t i = (size_t)blockIdx.x * blockDim.x + threadIdx.x;
  size_t total = (size_t)NWIN * NHEAD * 16 * 32;
  if (i >= total) return;
  int d = (int)(i & 31);
  size_t j = i >> 5;
  int jj = (int)(j % 16); j /= 16;
  int h = (int)(j % NHEAD);
  int b = (int)(j / NHEAD);
  float v = 0.f;
  if (jj < 4 && d < HD) v = kf[((size_t)b * 4 + jj) * DIMF + h * HD + d];
  dst[i] = f2bf(v);
}
// cross-attn V transposed: dst[(b*6+h)*32 + d][j<32] = vf[(b*4+j)*180 + h*30+d]
__global__ void cvt_cvt_k(const float* __restrict__ vf, unsigned short* __restrict__ dst) {
  size_t i = (size_t)blockIdx.x * blockDim.x + threadIdx.x;
  size_t total = (size_t)NWIN * NHEAD * 32 * 32;
  if (i >= total) return;
  int jj = (int)(i & 31);
  size_t j = i >> 5;
  int d = (int)(j % 32); j /= 32;
  int h = (int)(j % NHEAD);
  int b = (int)(j / NHEAD);
  float v = 0.f;
  if (jj < 4 && d < HD) v = vf[((size_t)b * 4 + jj) * DIMF + h * HD + d];
  dst[i] = f2bf(v);
}

// ---------------- windowed self-attention ----------------
// scores: one wave per 16x16 tile, single K=32 WMMA + relative-position bias
__global__ void sa_scores_k(const unsigned short* __restrict__ qb,
                            const unsigned short* __restrict__ kb,
                            const float* __restrict__ bias_tab,  // (529, 6)
                            float* __restrict__ S, int nb) {
  int lane = threadIdx.x & 31;
  int tile = blockIdx.x * (blockDim.x >> 5) + (threadIdx.x >> 5);
  if (tile >= nb * NHEAD * 81) return;
  int kt = tile % 9;
  int qt = (tile / 9) % 9;
  int bh = tile / 81;
  int half = lane >> 4, mn = lane & 15;
  const unsigned short* qp = qb + ((size_t)bh * NTOK + qt * 16 + mn) * 32;
  const unsigned short* kp = kb + ((size_t)bh * NTOK + kt * 16 + mn) * 32;
  f32x8 acc = {};
  acc = wmma_bf16(load_a_frag(qp, half), load_b_frag(kp, half), acc);
  int h = bh % NHEAD;
  int kpos = kt * 16 + mn;
  int ki = kpos / GSQ, kj = kpos % GSQ;
#pragma unroll
  for (int v = 0; v < 8; ++v) {
    int q = qt * 16 + v + 8 * half;
    int qi = q / GSQ, qj = q % GSQ;
    int idx = (qi - ki + GSQ - 1) * (2 * GSQ - 1) + (qj - kj + GSQ - 1);
    S[((size_t)bh * NTOK + q) * NTOK + kpos] = acc[v] + bias_tab[idx * NHEAD + h];
  }
}
// softmax over 144, wave per row, writes bf16 probabilities padded to 160
__global__ void sa_softmax_k(const float* __restrict__ S, unsigned short* __restrict__ pb,
                             int nrows) {
  int row = blockIdx.x * (blockDim.x >> 5) + (threadIdx.x >> 5);
  if (row >= nrows) return;
  int lane = threadIdx.x & 31;
  const float* sr = S + (size_t)row * NTOK;
  float vals[5];
  float mx = -1e30f;
#pragma unroll
  for (int i = 0; i < 5; ++i) {
    int c = lane + 32 * i;
    float v = (c < NTOK) ? sr[c] : -1e30f;
    vals[i] = v; mx = fmaxf(mx, v);
  }
#pragma unroll
  for (int m = 16; m; m >>= 1) mx = fmaxf(mx, __shfl_xor(mx, m, 32));
  float sum = 0.f;
#pragma unroll
  for (int i = 0; i < 5; ++i) {
    int c = lane + 32 * i;
    float e = (c < NTOK) ? __expf(vals[i] - mx) : 0.f;
    vals[i] = e; sum += e;
  }
#pragma unroll
  for (int m = 16; m; m >>= 1) sum += __shfl_xor(sum, m, 32);
  float inv = 1.f / sum;
  unsigned short* pr = pb + (size_t)row * KPAD;
#pragma unroll
  for (int i = 0; i < 5; ++i) {
    int c = lane + 32 * i;
    pr[c] = f2bf((c < NTOK) ? vals[i] * inv : 0.f);
  }
}
// AV: one wave per 16x16 output tile, 5 K-steps over padded 160
__global__ void sa_av_k(const unsigned short* __restrict__ pb,
                        const unsigned short* __restrict__ vt,
                        float* __restrict__ mout, int b0, int nb) {
  int lane = threadIdx.x & 31;
  int tile = blockIdx.x * (blockDim.x >> 5) + (threadIdx.x >> 5);
  if (tile >= nb * NHEAD * 18) return;
  int nt = tile % 2;
  int qt = (tile / 2) % 9;
  int bh = tile / 18;
  int half = lane >> 4, mn = lane & 15;
  const unsigned short* ap = pb + ((size_t)bh * NTOK + qt * 16 + mn) * KPAD;
  const unsigned short* bp = vt + ((size_t)bh * 32 + nt * 16 + mn) * KPAD;
  f32x8 acc = {};
#pragma unroll
  for (int kk = 0; kk < 5; ++kk)
    acc = wmma_bf16(load_a_frag(ap + kk * 32, half), load_b_frag(bp + kk * 32, half), acc);
  int h = bh % NHEAD;
  int wl = bh / NHEAD;
  int d = nt * 16 + mn;
  if (d < HD) {
#pragma unroll
    for (int v = 0; v < 8; ++v) {
      int q = qt * 16 + v + 8 * half;
      mout[((size_t)(b0 + wl) * NTOK + q) * DIMF + h * HD + d] = acc[v];
    }
  }
}

// ---------------- cross-attention (kv len 4) ----------------
__global__ void ca_scores_k(const unsigned short* __restrict__ cqb,
                            const unsigned short* __restrict__ ckb,
                            float* __restrict__ S2) {
  int lane = threadIdx.x & 31;
  int tile = blockIdx.x * (blockDim.x >> 5) + (threadIdx.x >> 5);
  if (tile >= NWIN * NHEAD * 9) return;
  int qt = tile % 9;
  int bh = tile / 9;
  int half = lane >> 4, mn = lane & 15;
  const unsigned short* qp = cqb + ((size_t)bh * NTOK + qt * 16 + mn) * 32;
  const unsigned short* kp = ckb + ((size_t)bh * 16 + mn) * 32;
  f32x8 acc = {};
  acc = wmma_bf16(load_a_frag(qp, half), load_b_frag(kp, half), acc);
  if (mn < 4) {
#pragma unroll
    for (int v = 0; v < 8; ++v) {
      int q = qt * 16 + v + 8 * half;
      S2[((size_t)bh * NTOK + q) * 4 + mn] = acc[v];
    }
  }
}
__global__ void ca_softmax_k(const float* __restrict__ S2, unsigned short* __restrict__ cpb) {
  size_t i = (size_t)blockIdx.x * blockDim.x + threadIdx.x;
  if (i >= (size_t)NWIN * NHEAD * NTOK) return;
  const float* s = S2 + i * 4;
  float m = fmaxf(fmaxf(s[0], s[1]), fmaxf(s[2], s[3]));
  float e0 = __expf(s[0] - m), e1 = __expf(s[1] - m);
  float e2 = __expf(s[2] - m), e3 = __expf(s[3] - m);
  float inv = 1.f / (e0 + e1 + e2 + e3);
  unsigned short* p = cpb + i * 32;
  p[0] = f2bf(e0 * inv); p[1] = f2bf(e1 * inv);
  p[2] = f2bf(e2 * inv); p[3] = f2bf(e3 * inv);
#pragma unroll
  for (int j = 4; j < 32; ++j) p[j] = 0;
}
__global__ void ca_av_k(const unsigned short* __restrict__ cpb,
                        const unsigned short* __restrict__ cvt2,
                        float* __restrict__ mout) {
  int lane = threadIdx.x & 31;
  int tile = blockIdx.x * (blockDim.x >> 5) + (threadIdx.x >> 5);
  if (tile >= NWIN * NHEAD * 18) return;
  int nt = tile % 2;
  int qt = (tile / 2) % 9;
  int bh = tile / 18;
  int half = lane >> 4, mn = lane & 15;
  const unsigned short* ap = cpb + ((size_t)bh * NTOK + qt * 16 + mn) * 32;
  const unsigned short* bp = cvt2 + ((size_t)bh * 32 + nt * 16 + mn) * 32;
  f32x8 acc = {};
  acc = wmma_bf16(load_a_frag(ap, half), load_b_frag(bp, half), acc);
  int h = bh % NHEAD;
  int wl = bh / NHEAD;
  int d = nt * 16 + mn;
  if (d < HD) {
#pragma unroll
    for (int v = 0; v < 8; ++v) {
      int q = qt * 16 + v + 8 * half;
      mout[((size_t)wl * NTOK + q) * DIMF + h * HD + d] = acc[v];
    }
  }
}

// ---------------- window roll (+ optional residual) ----------------
__global__ void roll_add_k(const float* __restrict__ src, const float* __restrict__ res,
                           float* __restrict__ dst, int shift) {
  size_t i = (size_t)blockIdx.x * blockDim.x + threadIdx.x;
  const size_t total = (size_t)NWIN * NTOK * DIMF;
  if (i >= total) return;
  int c = (int)(i % DIMF);
  size_t j = i / DIMF;
  int t = (int)(j % NTOK);
  size_t bw = j / NTOK;
  int n = (int)(bw % GSQ);
  size_t tmp = bw / GSQ;
  int mw = (int)(tmp % GSQ);
  int b = (int)(tmp / GSQ);
  int ti = t / GSQ, tj = t % GSQ;
  int gi = mw * GSQ + ti, gj = n * GSQ + tj;
  int sgi = (gi - shift) % SIDE; if (sgi < 0) sgi += SIDE;
  int sgj = (gj - shift) % SIDE; if (sgj < 0) sgj += SIDE;
  int sm = sgi / GSQ, si = sgi % GSQ, sn = sgj / GSQ, sj = sgj % GSQ;
  size_t sidx = (((size_t)b * (GSQ * GSQ) + sm * GSQ + sn) * NTOK + si * GSQ + sj) * DIMF + c;
  float v = src[sidx];
  if (res) v += res[i];
  dst[i] = v;
}

// ---------------- host orchestration ----------------
static inline unsigned cdiv(size_t a, size_t b) { return (unsigned)((a + b - 1) / b); }

extern "C" void kernel_launch(void* const* d_in, const int* in_sizes, int n_in,
                              void* d_out, int out_size, void* d_ws, size_t ws_size,
                              hipStream_t stream) {
  (void)in_sizes; (void)n_in; (void)out_size; (void)ws_size;
  const float* gs        = (const float*)d_in[0];
  const float* pos       = (const float*)d_in[1];
  const float* sce       = (const float*)d_in[2];
  const float* blknw     = (const float*)d_in[3];
  const float* blknb     = (const float*)d_in[4];
  const float* blk1w     = (const float*)d_in[5];
  const float* blk1b     = (const float*)d_in[6];
  const float* blk2w     = (const float*)d_in[7];
  const float* blk2b     = (const float*)d_in[8];
  const float* sa_qw     = (const float*)d_in[9];
  const float* sa_qb     = (const float*)d_in[10];
  const float* sa_kw     = (const float*)d_in[11];
  const float* sa_kb     = (const float*)d_in[12];
  const float* sa_vw     = (const float*)d_in[13];
  const float* sa_vb     = (const float*)d_in[14];
  const float* sa_pw     = (const float*)d_in[15];
  const float* sa_pb     = (const float*)d_in[16];
  const float* sa_bias   = (const float*)d_in[17];
  const float* ca_inw    = (const float*)d_in[18];
  const float* ca_inb    = (const float*)d_in[19];
  const float* ca_ow     = (const float*)d_in[20];
  const float* ca_ob     = (const float*)d_in[21];
  const float* msa1w     = (const float*)d_in[22];
  const float* msa1b     = (const float*)d_in[23];
  const float* msa2w     = (const float*)d_in[24];
  const float* msa2b     = (const float*)d_in[25];
  const float* mca1w     = (const float*)d_in[26];
  const float* mca1b     = (const float*)d_in[27];
  const float* mca2w     = (const float*)d_in[28];
  const float* mca2b     = (const float*)d_in[29];
  const float* ln1w = (const float*)d_in[30];
  const float* ln1b = (const float*)d_in[31];
  const float* ln2w = (const float*)d_in[32];
  const float* ln2b = (const float*)d_in[33];
  const float* ln3w = (const float*)d_in[34];
  const float* ln3b = (const float*)d_in[35];
  const float* ln4w = (const float*)d_in[36];
  const float* ln4b = (const float*)d_in[37];
  float* outp = (float*)d_out;

  // scratch carve
  char* wsp = (char*)d_ws;
  auto alloc = [&](size_t bytes) -> void* {
    void* p = (void*)wsp;
    wsp += (bytes + 255) & ~(size_t)255;
    return p;
  };
  const size_t FMB = (size_t)MTOT * DIMF * sizeof(float);
  float* x  = (float*)alloc(FMB);
  float* y  = (float*)alloc(FMB);
  float* z  = (float*)alloc(FMB);
  float* g1 = (float*)alloc(FMB);
  float* g2 = (float*)alloc(FMB);
  float* g3 = (float*)alloc(FMB);
  float* mg = (float*)alloc(FMB);
  unsigned short* abf  = (unsigned short*)alloc((size_t)MTOT * DP * 2);
  unsigned short* abf2 = (unsigned short*)alloc((size_t)MTOT * DP * 2);
  unsigned short* sbf  = (unsigned short*)alloc((size_t)MKV * DP * 2);
  unsigned short* wbf  = (unsigned short*)alloc((size_t)DP * DP * 2);
  unsigned short* cqb  = (unsigned short*)alloc((size_t)NWIN * NHEAD * NTOK * 32 * 2);
  unsigned short* ckb  = (unsigned short*)alloc((size_t)NWIN * NHEAD * 16 * 32 * 2);
  unsigned short* cvt2 = (unsigned short*)alloc((size_t)NWIN * NHEAD * 32 * 32 * 2);
  float*          S2   = (float*)alloc((size_t)NWIN * NHEAD * NTOK * 4 * sizeof(float));
  unsigned short* cpb  = (unsigned short*)alloc((size_t)NWIN * NHEAD * NTOK * 32 * 2);
  unsigned short* qb   = (unsigned short*)alloc((size_t)CH * NHEAD * NTOK * 32 * 2);
  unsigned short* kb   = (unsigned short*)alloc((size_t)CH * NHEAD * NTOK * 32 * 2);
  unsigned short* vt   = (unsigned short*)alloc((size_t)CH * NHEAD * 32 * KPAD * 2);
  float*          S    = (float*)alloc((size_t)CH * NHEAD * NTOK * NTOK * sizeof(float));
  unsigned short* pb   = (unsigned short*)alloc((size_t)CH * NHEAD * NTOK * KPAD * 2);

  const float qscale = 0.18257418583505536f; // 1/sqrt(30)

  auto ln = [&](const float* src, const float* w, const float* b, const float* add,
                float* dst, int rows) {
    ln_k<<<cdiv(rows, 8), 256, 0, stream>>>(src, w, b, add, dst, rows);
  };
  auto cvtA = [&](const float* src, unsigned short* dst, int rows) {
    cvtA_k<<<cdiv((size_t)rows * DP, 256), 256, 0, stream>>>(src, dst, rows);
  };
  auto gemm = [&](const unsigned short* A, const float* Wsrc, const float* bias,
                  const float* res, float* C, unsigned short* Cbf, int rows, int relu) {
    cvtW_k<<<cdiv(DP * DP, 256), 256, 0, stream>>>(Wsrc, wbf);
    gemm_wmma<<<cdiv((size_t)(rows / 16) * 3, 8), 256, 0, stream>>>(A, wbf, bias, res, C,
                                                                    Cbf, rows, relu);
  };

  // x = LN(gs) ; staging of scale embedding (constant across layers)
  ln(gs, blknw, blknb, nullptr, x, MTOT);
  cvtA(sce, sbf, MKV);

  for (int l = 0; l < 4; ++l) {
    const int shift = (l & 1) ? SHIFT : 0;
    const float* wq = ca_inw + (size_t)l * 3 * DIMF * DIMF;
    const float* wk = wq + DIMF * DIMF;
    const float* wv = wk + DIMF * DIMF;
    const float* bq = ca_inb + (size_t)l * 3 * DIMF;
    const float* bk = bq + DIMF;
    const float* bv = bk + DIMF;

    // ---- cross-attention to scale embedding ----
    ln(x, ln3w + l * DIMF, ln3b + l * DIMF, pos, y, MTOT);
    cvtA(y, abf, MTOT);
    gemm(abf, wq, bq, nullptr, g1, nullptr, MTOT, 0);
    gemm(sbf, wk, bk, nullptr, g2, nullptr, MKV, 0);
    gemm(sbf, wv, bv, nullptr, g3, nullptr, MKV, 0);
    cvt_head_k<<<cdiv((size_t)NWIN * NHEAD * NTOK * 32, 256), 256, 0, stream>>>(
        g1, cqb, qscale, NWIN, 0);
    cvt_ckb_k<<<cdiv((size_t)NWIN * NHEAD * 16 * 32, 256), 256, 0, stream>>>(g2, ckb);
    cvt_cvt_k<<<cdiv((size_t)NWIN * NHEAD * 32 * 32, 256), 256, 0, stream>>>(g3, cvt2);
    ca_scores_k<<<cdiv((size_t)NWIN * NHEAD * 9, 8), 256, 0, stream>>>(cqb, ckb, S2);
    ca_softmax_k<<<cdiv((size_t)NWIN * NHEAD * NTOK, 256), 256, 0, stream>>>(S2, cpb);
    ca_av_k<<<cdiv((size_t)NWIN * NHEAD * 18, 8), 256, 0, stream>>>(cpb, cvt2, y);
    cvtA(y, abf, MTOT);
    gemm(abf, ca_ow + (size_t)l * DIMF * DIMF, ca_ob + l * DIMF, x, x, nullptr, MTOT, 0);

    // ---- cross MLP (hidden activation stays bf16, no f32 round-trip) ----
    ln(x, ln4w + l * DIMF, ln4b + l * DIMF, nullptr, y, MTOT);
    cvtA(y, abf, MTOT);
    gemm(abf, mca1w + (size_t)l * DIMF * DIMF, mca1b + l * DIMF, nullptr, nullptr, abf2,
         MTOT, 1);
    gemm(abf2, mca2w + (size_t)l * DIMF * DIMF, mca2b + l * DIMF, x, x, nullptr, MTOT, 0);

    // ---- (shifted) windowed self-attention ----
    ln(x, ln1w + l * DIMF, ln1b + l * DIMF, nullptr, y, MTOT);
    const float* sain = y;
    if (shift) {
      roll_add_k<<<cdiv((size_t)MTOT * DIMF, 256), 256, 0, stream>>>(y, nullptr, z, -SHIFT);
      sain = z;
    }
    cvtA(sain, abf, MTOT);
    gemm(abf, sa_qw + (size_t)l * DIMF * DIMF, sa_qb + l * DIMF, nullptr, g1, nullptr,
         MTOT, 0);
    gemm(abf, sa_kw + (size_t)l * DIMF * DIMF, sa_kb + l * DIMF, nullptr, g2, nullptr,
         MTOT, 0);
    gemm(abf, sa_vw + (size_t)l * DIMF * DIMF, sa_vb + l * DIMF, nullptr, g3, nullptr,
         MTOT, 0);
    const float* btab = sa_bias + (size_t)l * (2 * GSQ - 1) * (2 * GSQ - 1) * NHEAD;
    for (int c = 0; c < NCHUNK; ++c) {
      int b0 = c * CH;
      cvt_head_k<<<cdiv((size_t)CH * NHEAD * NTOK * 32, 256), 256, 0, stream>>>(
          g1, qb, qscale, CH, b0);
      cvt_head_k<<<cdiv((size_t)CH * NHEAD * NTOK * 32, 256), 256, 0, stream>>>(
          g2, kb, 1.0f, CH, b0);
      cvt_vt_k<<<cdiv((size_t)CH * NHEAD * 32 * KPAD, 256), 256, 0, stream>>>(g3, vt, CH, b0);
      sa_scores_k<<<cdiv((size_t)CH * NHEAD * 81, 8), 256, 0, stream>>>(qb, kb, btab, S, CH);
      sa_softmax_k<<<cdiv((size_t)CH * NHEAD * NTOK, 8), 256, 0, stream>>>(
          S, pb, CH * NHEAD * NTOK);
      sa_av_k<<<cdiv((size_t)CH * NHEAD * 18, 8), 256, 0, stream>>>(pb, vt, mg, b0, CH);
    }
    cvtA(mg, abf, MTOT);
    if (shift) {
      gemm(abf, sa_pw + (size_t)l * DIMF * DIMF, sa_pb + l * DIMF, nullptr, z, nullptr,
           MTOT, 0);
      roll_add_k<<<cdiv((size_t)MTOT * DIMF, 256), 256, 0, stream>>>(z, x, x, SHIFT);
    } else {
      gemm(abf, sa_pw + (size_t)l * DIMF * DIMF, sa_pb + l * DIMF, x, x, nullptr, MTOT, 0);
    }

    // ---- self MLP (hidden activation stays bf16) ----
    ln(x, ln2w + l * DIMF, ln2b + l * DIMF, nullptr, y, MTOT);
    cvtA(y, abf, MTOT);
    gemm(abf, msa1w + (size_t)l * DIMF * DIMF, msa1b + l * DIMF, nullptr, nullptr, abf2,
         MTOT, 1);
    gemm(abf2, msa2w + (size_t)l * DIMF * DIMF, msa2b + l * DIMF, x, x, nullptr, MTOT, 0);
  }

  // ---- final block MLP + residual(gs) -> d_out ----
  cvtA(x, abf, MTOT);
  gemm(abf, blk1w, blk1b, nullptr, nullptr, abf2, MTOT, 1);
  gemm(abf2, blk2w, blk2b, gs, outp, nullptr, MTOT, 0);
}
